// OffGridPatchEmbedGather_1297080123683
// MI455X (gfx1250) — matrix operand order repaired and checked by reading, not compiled, not run on hardware
//
#include <hip/hip_runtime.h>
#include <hip/hip_bf16.h>

typedef __attribute__((ext_vector_type(16))) _Float16 v16h;
typedef __attribute__((ext_vector_type(8)))  float    v8f;

#define IMGW 224
#define PCH  16
#define NCH  3
#define NTOK 196
#define KDIM 768     // 3*16*16
#define DDIM 384
#define BATCH 128
#define KSTEPS 24    // 768 / 32
#define CTILES 24    // 384 / 16
#define MTILES 2     // 16-row tiles per workgroup
#define WGROWS (MTILES * 16)      // 32 tokens per workgroup
#define WGS_PER_B 7               // ceil(196 / 32) -> rows 0..223, padded

// ---------------------------------------------------------------------------
// Pack W [768 x 384] fp32 -> f16 in WMMA B-operand order.
// B operand (32x16 f16, K x N): lane L holds column N = L&15; half = L>>4;
// element j (0..15) holds K = j + 16*half  (contiguous k order used by this code,
// same as A; lanes 16-31 carry the upper K half).
// Layout: Wp[((tile*KSTEPS + s)*32 + lane)*16 + j]
// ---------------------------------------------------------------------------
__global__ __launch_bounds__(256) void packw_kernel(const float* __restrict__ W,
                                                    _Float16* __restrict__ Wp) {
    int idx = blockIdx.x * 256 + threadIdx.x;       // (tile, s, lane)
    if (idx >= CTILES * KSTEPS * 32) return;
    int lane = idx & 31;
    int s    = (idx >> 5) % KSTEPS;
    int tile = idx / (KSTEPS * 32);
    int half = lane >> 4;
    int col  = tile * 16 + (lane & 15);
    _Float16* dst = Wp + (size_t)idx * 16;
    #pragma unroll
    for (int j = 0; j < 16; ++j) {
        int K = s * 32 + j + 16 * half;
        dst[j] = (_Float16)W[(size_t)K * DDIM + col];
    }
}

// ---------------------------------------------------------------------------
// positions = stack([ys, xs], -1)  (int32 tail of d_out)
// ---------------------------------------------------------------------------
__global__ __launch_bounds__(256) void pos_kernel(const int* __restrict__ ys,
                                                  const int* __restrict__ xs,
                                                  int* __restrict__ outp) {
    int i = blockIdx.x * 256 + threadIdx.x;
    if (i < BATCH * NTOK) {
        outp[2 * i + 0] = ys[i];
        outp[2 * i + 1] = xs[i];
    }
}

// ---------------------------------------------------------------------------
// Fused gather + GEMM. One workgroup = (batch, 32-token slab = 2 M-tiles).
// 8 waves; wave w owns column tiles 3w..3w+2 and BOTH M-tiles, so each
// B-operand register pair feeds two v_wmma issues (halves L2 W-traffic).
// A staged in LDS in WMMA A-operand order (16-bit A 16x32, this code's contiguous k order):
//   lane L: M = L&15, half = L>>4; element j: K = j + 16*half
// ---------------------------------------------------------------------------
__global__ __launch_bounds__(256) void gemm_kernel(const float* __restrict__ x,
                                                   const int* __restrict__ ys,
                                                   const int* __restrict__ xs,
                                                   const _Float16* __restrict__ Wp,
                                                   const float* __restrict__ bias,
                                                   float* __restrict__ tokens) {
    __shared__ _Float16 Apack[MTILES][KSTEPS * 32 * 16];   // 2 x 24 KB
    __shared__ int      baseoff[WGROWS];

    const int b    = blockIdx.x / WGS_PER_B;
    const int slab = blockIdx.x % WGS_PER_B;
    const int row0 = slab * WGROWS;                 // first token of this WG
    const int tid  = threadIdx.x;

    // Per-row patch base offsets (y*224 + x); -1 => padded row (zeros).
    if (tid < WGROWS) {
        int n = row0 + tid;
        int off = -1;
        if (n < NTOK) {
            off = ys[b * NTOK + n] * IMGW + xs[b * NTOK + n];
        }
        baseoff[tid] = off;
    }
    __syncthreads();

    // -------- Phase 1: gather patches -> LDS (A-operand layout, f16) --------
    const float* xb = x + (size_t)b * NCH * IMGW * IMGW;
    for (int i = tid; i < WGROWS * KDIM; i += 256) {
        int row = i / KDIM;                // 0..31
        int k   = i - row * KDIM;          // 0..767 = c*256 + dy*16 + dx
        int off = baseoff[row];
        float v = 0.0f;
        if (off >= 0) {
            int c   = k >> 8;
            int rem = k & 255;
            int dy  = rem >> 4;
            int dx  = rem & 15;
            v = xb[c * (IMGW * IMGW) + off + dy * IMGW + dx];
        }
        int mt    = row >> 4;              // which M-tile
        int rowin = row & 15;
        int s     = k >> 5;
        int kk    = k & 31;
        int half  = kk >> 4;
        int j     = kk & 15;
        int L     = rowin + 16 * half;
        Apack[mt][(s * 32 + L) * 16 + j] = (_Float16)v;
    }
    __syncthreads();

    // -------- Phase 2: WMMA GEMM, 2 M-tiles x 3 N-tiles per wave --------
    const int wave = tid >> 5;
    const int lane = tid & 31;
    const int t0   = wave * 3;

    v8f acc00 = {}, acc01 = {}, acc02 = {};   // M-tile 0
    v8f acc10 = {}, acc11 = {}, acc12 = {};   // M-tile 1

    const _Float16* wp0 = Wp + (size_t)((t0 + 0) * KSTEPS) * 32 * 16;
    const _Float16* wp1 = Wp + (size_t)((t0 + 1) * KSTEPS) * 32 * 16;
    const _Float16* wp2 = Wp + (size_t)((t0 + 2) * KSTEPS) * 32 * 16;

    for (int s = 0; s < KSTEPS; ++s) {
        v16h a0 = *(const v16h*)&Apack[0][(s * 32 + lane) * 16];
        v16h a1 = *(const v16h*)&Apack[1][(s * 32 + lane) * 16];
        v16h b0 = *(const v16h*)&wp0[(s * 32 + lane) * 16];
        v16h b1 = *(const v16h*)&wp1[(s * 32 + lane) * 16];
        v16h b2 = *(const v16h*)&wp2[(s * 32 + lane) * 16];
        acc00 = __builtin_amdgcn_wmma_f32_16x16x32_f16(false, a0, false, b0,
                                                       (short)0, acc00, false, false);
        acc01 = __builtin_amdgcn_wmma_f32_16x16x32_f16(false, a0, false, b1,
                                                       (short)0, acc01, false, false);
        acc02 = __builtin_amdgcn_wmma_f32_16x16x32_f16(false, a0, false, b2,
                                                       (short)0, acc02, false, false);
        acc10 = __builtin_amdgcn_wmma_f32_16x16x32_f16(false, a1, false, b0,
                                                       (short)0, acc10, false, false);
        acc11 = __builtin_amdgcn_wmma_f32_16x16x32_f16(false, a1, false, b1,
                                                       (short)0, acc11, false, false);
        acc12 = __builtin_amdgcn_wmma_f32_16x16x32_f16(false, a1, false, b2,
                                                       (short)0, acc12, false, false);
    }

    // -------- Epilogue: bias + store (C/D layout: elem v -> M = v + 8*half) --
    const int half = lane >> 4;
    const int ncol = lane & 15;

    v8f accs[2][3] = {{acc00, acc01, acc02}, {acc10, acc11, acc12}};

    #pragma unroll
    for (int mt = 0; mt < MTILES; ++mt) {
        #pragma unroll
        for (int ti = 0; ti < 3; ++ti) {
            int col = (t0 + ti) * 16 + ncol;
            float bc = bias[col];
            v8f acc = accs[mt][ti];
            #pragma unroll
            for (int v = 0; v < 8; ++v) {
                int n = row0 + mt * 16 + v + 8 * half;
                if (n < NTOK) {
                    tokens[((size_t)b * NTOK + n) * DDIM + col] = acc[v] + bc;
                }
            }
        }
    }
}

extern "C" void kernel_launch(void* const* d_in, const int* in_sizes, int n_in,
                              void* d_out, int out_size, void* d_ws, size_t ws_size,
                              hipStream_t stream) {
    const float* x    = (const float*)d_in[0];
    const int*   ys   = (const int*)d_in[1];
    const int*   xs   = (const int*)d_in[2];
    const float* W    = (const float*)d_in[3];
    const float* bias = (const float*)d_in[4];

    float* tokens    = (float*)d_out;
    int*   positions = (int*)d_out + (size_t)BATCH * NTOK * DDIM;

    _Float16* Wp = (_Float16*)d_ws;   // 24*24*32*16 f16 = 576 KB

    int packElems = CTILES * KSTEPS * 32;
    packw_kernel<<<(packElems + 255) / 256, 256, 0, stream>>>(W, Wp);
    pos_kernel<<<(BATCH * NTOK + 255) / 256, 256, 0, stream>>>(ys, xs, positions);
    gemm_kernel<<<BATCH * WGS_PER_B, 256, 0, stream>>>(x, ys, xs, Wp, bias, tokens);
}
